// BitGenModel_43714177139359
// MI455X (gfx1250) — 8-bit matrix operand order repaired and checked by reading, compiled, not run on hardware
//
#include <hip/hip_runtime.h>
#include <stdint.h>


typedef __attribute__((ext_vector_type(8))) int v8i;

namespace {

constexpr int kB = 8, kS = 256, kE = 1024, kH = 16, kD = 64, kL = 8;
constexpr int kFF = 4096, kV = 8192, kVD = 1024;
constexpr int kPatchK  = 3 * 14 * 14;
constexpr int kPatchKp = 640;
constexpr int kM   = kB * kS;
constexpr int kMF  = kB * (kS + 1);
constexpr int kMFp = 2112;
constexpr int kMV  = 64;

constexpr size_t align_up(size_t x) { return (x + 255) & ~(size_t)255; }

constexpr size_t OFF_SC = 0;
constexpr size_t OFF_QX = 4096;
constexpr size_t SZ_QX  = align_up((size_t)kMFp * kFF);
constexpr size_t OFF_QW = OFF_QX + SZ_QX;
constexpr size_t SZ_QW  = align_up((size_t)kV * kE);
constexpr size_t OFF_X  = OFF_QW + SZ_QW;
constexpr size_t SZ_X   = align_up((size_t)kM * kE * 4);
constexpr size_t OFF_H  = OFF_X + SZ_X;
constexpr size_t SZ_H   = align_up((size_t)kM * kFF * 4);
constexpr size_t OFF_FU = OFF_H + SZ_H;
constexpr size_t SZ_FU  = align_up((size_t)kM * 2 * kE * 4);
constexpr size_t SZ_ACT = align_up((size_t)kMF * kE * 4);
constexpr size_t OFF_Q  = OFF_FU + SZ_FU;
constexpr size_t OFF_K  = OFF_Q + SZ_ACT;
constexpr size_t OFF_V  = OFF_K + SZ_ACT;
constexpr size_t OFF_AO = OFF_V + SZ_ACT;
constexpr size_t OFF_XC = OFF_AO + SZ_ACT;
constexpr size_t OFF_VP = OFF_XC + SZ_ACT;
constexpr size_t SZ_VP  = align_up((size_t)kB * kPatchK * 4);
constexpr size_t OFF_VH = OFF_VP + SZ_VP;
constexpr size_t SZ_VH  = align_up((size_t)kB * kVD * 4);
constexpr size_t OFF_VO = OFF_VH + SZ_VH;

}


__global__ void k_embed(const int* __restrict__ ids, const float* __restrict__ emb,
                        float* __restrict__ x) {
  int i = blockIdx.x * blockDim.x + threadIdx.x;
  if (i >= kM * kE) return;
  int m = i >> 10, e = i & (kE - 1);
  x[i] = emb[(size_t)ids[m] * kE + e];
}

__global__ void k_pool(const float* __restrict__ img, float* __restrict__ patch) {
  int i = blockIdx.x * blockDim.x + threadIdx.x;
  if (i >= kB * kPatchK) return;
  int b = i / kPatchK, r = i % kPatchK;
  int c = r / 196, pr = r % 196, py = pr / 14, px = pr % 14;
  const float* base = img + (((size_t)b * 3 + c) * 224 + py * 16) * 224 + px * 16;
  float s = 0.f;
  for (int y = 0; y < 16; ++y)
    for (int x2 = 0; x2 < 16; ++x2) s += base[y * 224 + x2];
  patch[i] = s * (1.0f / 256.0f);
}

__global__ void k_build_xcat(const float* __restrict__ x, const float* __restrict__ vis,
                             float* __restrict__ xcat) {
  int i = blockIdx.x * blockDim.x + threadIdx.x;
  if (i >= kMF * kE) return;
  int m = i >> 10, e = i & (kE - 1);
  int b = m / (kS + 1), s = m % (kS + 1);
  xcat[i] = (s < kS) ? x[((size_t)b * kS + s) * kE + e] : vis[(size_t)b * kE + e];
}

__global__ void k_concat_fused(const float* __restrict__ x, const float* __restrict__ attp,
                               float* __restrict__ fused) {
  int i = blockIdx.x * blockDim.x + threadIdx.x;
  if (i >= kM * 2 * kE) return;
  int m = i >> 11, e = i & (2 * kE - 1);
  if (e < kE) fused[i] = x[(size_t)m * kE + e];
  else {
    int b = m / kS, s = m % kS;
    fused[i] = attp[((size_t)b * (kS + 1) + s) * kE + (e - kE)];
  }
}

__global__ void k_add(float* __restrict__ x, const float* __restrict__ a, int n) {
  int i = blockIdx.x * blockDim.x + threadIdx.x;
  if (i < n) x[i] += a[i];
}


__global__ void k_part_absmax(const float* __restrict__ x, int n, float* __restrict__ part) {
  __shared__ float red[256];
  float m = 0.f;
  for (int i = blockIdx.x * 256 + threadIdx.x; i < n; i += 256 * 256)
    m = fmaxf(m, fabsf(x[i]));
  red[threadIdx.x] = m; __syncthreads();
  for (int s = 128; s > 0; s >>= 1) {
    if (threadIdx.x < s) red[threadIdx.x] = fmaxf(red[threadIdx.x], red[threadIdx.x + s]);
    __syncthreads();
  }
  if (threadIdx.x == 0) part[blockIdx.x] = red[0];
}

__global__ void k_finish_max(const float* __restrict__ part, float* __restrict__ scal) {
  __shared__ float red[256];
  red[threadIdx.x] = part[threadIdx.x]; __syncthreads();
  for (int s = 128; s > 0; s >>= 1) {
    if (threadIdx.x < s) red[threadIdx.x] = fmaxf(red[threadIdx.x], red[threadIdx.x + s]);
    __syncthreads();
  }
  if (threadIdx.x == 0) scal[0] = red[0];
}

__global__ void k_part_abssum(const float* __restrict__ x, int n, float* __restrict__ part) {
  __shared__ float red[256];
  float a = 0.f;
  for (int i = blockIdx.x * 256 + threadIdx.x; i < n; i += 256 * 256)
    a += fabsf(x[i]);
  red[threadIdx.x] = a; __syncthreads();
  for (int s = 128; s > 0; s >>= 1) {
    if (threadIdx.x < s) red[threadIdx.x] += red[threadIdx.x + s];
    __syncthreads();
  }
  if (threadIdx.x == 0) part[blockIdx.x] = red[0];
}

__global__ void k_finish_sum(const float* __restrict__ part, float* __restrict__ scal) {
  __shared__ float red[256];
  red[threadIdx.x] = part[threadIdx.x]; __syncthreads();
  for (int s = 128; s > 0; s >>= 1) {
    if (threadIdx.x < s) red[threadIdx.x] += red[threadIdx.x + s];
    __syncthreads();
  }
  if (threadIdx.x == 0) scal[0] = red[0];
}


__global__ void k_quant_x(const float* __restrict__ x, int8_t* __restrict__ qx,
                          const float* __restrict__ amax, int M, int K, int Mp, int Kp) {
  int i = blockIdx.x * blockDim.x + threadIdx.x;
  if (i >= Mp * Kp) return;
  int m = i / Kp, c = i % Kp;
  int8_t q = 0;
  if (m < M && c < K) {
    float isc = fmaxf(amax[0] * (1.0f / 127.0f), 1e-30f);
    float r = rintf(x[(size_t)m * K + c] / isc);
    r = fminf(fmaxf(r, -128.f), 127.f);
    q = (int8_t)(int)r;
  }
  qx[i] = q;
}

__global__ void k_quant_w(const float* __restrict__ w, int8_t* __restrict__ qw,
                          const float* __restrict__ wsum, int N, int K, int Kp) {
  int i = blockIdx.x * blockDim.x + threadIdx.x;
  if (i >= N * Kp) return;
  int n = i / Kp, c = i % Kp;
  int8_t q = 0;
  if (c < K) {
    float thr = 0.5f * wsum[0] / (float)((long)N * K);
    float wv = w[(size_t)n * K + c];
    if (fabsf(wv) > thr) q = (wv > 0.f) ? 1 : -1;
  }
  qw[i] = q;
}


__device__ __forceinline__ v8i load_afrag(const int8_t* __restrict__ arow,
                                          int kk, int hi) {
  v8i a;
#pragma unroll
  for (int r = 0; r < 8; ++r)
    a[r] = *(const int*)(arow + kk + 32 * (r >> 2) + 16 * hi + 4 * (r & 3));
  return a;
}

__device__ __forceinline__ v8i load_bfrag(const int8_t* __restrict__ bbase,
                                          int ns, int lm, int hi) {
  const int8_t* brow = bbase + (ns * 16 + lm) * 64;
  const int4 blo = *(const int4*)(brow + 16 * hi);
  const int4 bhi = *(const int4*)(brow + 32 + 16 * hi);
  return v8i{blo.x, blo.y, blo.z, blo.w, bhi.x, bhi.y, bhi.z, bhi.w};
}

__global__ void __launch_bounds__(128)
k_gemm_i8(const int8_t* __restrict__ qx, const int8_t* __restrict__ qw,
          const float* __restrict__ bias, const float* __restrict__ wsum,
          const float* __restrict__ amax, float* __restrict__ out,
          int Mvalid, int N, int Kp, int wNK, int relu) {
  __shared__ alignas(16) int8_t lb[2][64 * 64];

  const int t    = threadIdx.x;
  const int lane = t & 31;
  const int wv   = t >> 5;
  const int m0 = blockIdx.x * 64;
  const int n0 = blockIdx.y * 64;
  const int hi = lane >> 4;
  const int lm = lane & 15;
  const int msub = m0 + wv * 16;

  v8i acc[4];
#pragma unroll
  for (int ns = 0; ns < 4; ++ns) acc[ns] = v8i{0, 0, 0, 0, 0, 0, 0, 0};

  const int8_t* arow = qx + (size_t)(msub + lm) * Kp;
  const uint32_t lbase = (uint32_t)(uintptr_t)&lb[0][0];

  auto stage = [&](int kk, int buf) {
#pragma unroll
    for (int c = 0; c < 2; ++c) {
      int ch  = t + c * 128;
      int n   = ch >> 2;
      int col = (ch & 3) * 16;
      unsigned long long g =
          (unsigned long long)(const void*)(qw + (size_t)(n0 + n) * Kp + kk + col);
      uint32_t l = lbase + (uint32_t)(buf * 4096 + n * 64 + col);
      asm volatile("global_load_async_to_lds_b128 %0, %1, off"
                   :: "v"(l), "v"(g) : "memory");
    }
  };

  const int nsteps = Kp >> 6;
  stage(0, 0);
  v8i a_cur = load_afrag(arow, 0, hi);

  for (int s = 0; s < nsteps; ++s) {
    asm volatile("s_wait_asynccnt 0x0" ::: "memory");
    __syncthreads();
    const bool more = (s + 1 < nsteps);
    if (more) stage((s + 1) << 6, (s + 1) & 1);

    const int8_t* bbase = &lb[s & 1][0];
    v8i b0 = load_bfrag(bbase, 0, lm, hi);
    v8i b1 = load_bfrag(bbase, 1, lm, hi);
    v8i b2 = load_bfrag(bbase, 2, lm, hi);
    v8i b3 = load_bfrag(bbase, 3, lm, hi);

    v8i a_next = a_cur;
    if (more) a_next = load_afrag(arow, (s + 1) << 6, hi);

    acc[0] = __builtin_amdgcn_wmma_i32_16x16x64_iu8(true, a_cur, true, b0, acc[0], false, false);
    acc[1] = __builtin_amdgcn_wmma_i32_16x16x64_iu8(true, a_cur, true, b1, acc[1], false, false);
    acc[2] = __builtin_amdgcn_wmma_i32_16x16x64_iu8(true, a_cur, true, b2, acc[2], false, false);
    acc[3] = __builtin_amdgcn_wmma_i32_16x16x64_iu8(true, a_cur, true, b3, acc[3], false, false);

    __syncthreads();
    a_cur = a_next;
  }

  const float scale = (wsum[0] / (float)wNK) * (amax[0] * (1.0f / 127.0f));
#pragma unroll
  for (int ns = 0; ns < 4; ++ns) {
    const int n = n0 + ns * 16 + lm;
    const float bn = bias[n];
#pragma unroll
    for (int r = 0; r < 8; ++r) {
      int m = msub + r + 8 * hi;
      if (m < Mvalid) {
        float vout = (float)acc[ns][r] * scale + bn;
        if (relu) vout = fmaxf(vout, 0.f);
        out[(size_t)m * N + n] = vout;
      }
    }
  }
}


__global__ void __launch_bounds__(256)
k_attn(const float* __restrict__ q, const float* __restrict__ k,
       const float* __restrict__ v, float* __restrict__ o, int Srun) {
  const int qi = blockIdx.x, h = blockIdx.y, b = blockIdx.z;
  const int tid = threadIdx.x;
  __shared__ float sc[kS + 4];
  __shared__ float red[256];
  __shared__ float qv[kD];
  __shared__ float ph[256];

  const size_t rowq = ((size_t)b * Srun + qi) * kE + h * kD;
  if (tid < kD) qv[tid] = q[rowq + tid];
  __syncthreads();

  float lmax = -3.0e38f;
  for (int kj = tid; kj <= qi; kj += 256) {
    const float* kr = k + ((size_t)b * Srun + kj) * kE + h * kD;
    float dot = 0.f;
#pragma unroll
    for (int d = 0; d < kD; ++d) dot += qv[d] * kr[d];
    dot *= 0.125f;
    sc[kj] = dot;
    lmax = fmaxf(lmax, dot);
  }
  red[tid] = lmax; __syncthreads();
  for (int s = 128; s > 0; s >>= 1) {
    if (tid < s) red[tid] = fmaxf(red[tid], red[tid + s]);
    __syncthreads();
  }
  const float rmax = red[0]; __syncthreads();

  float lsum = 0.f;
  for (int kj = tid; kj <= qi; kj += 256) {
    float e = __expf(sc[kj] - rmax);
    sc[kj] = e; lsum += e;
  }
  red[tid] = lsum; __syncthreads();
  for (int s = 128; s > 0; s >>= 1) {
    if (tid < s) red[tid] += red[tid + s];
    __syncthreads();
  }
  const float inv = 1.0f / red[0]; __syncthreads();

  const int d = tid & 63, ch = tid >> 6;
  float acc = 0.f;
  for (int kj = ch; kj <= qi; kj += 4)
    acc += sc[kj] * v[((size_t)b * Srun + kj) * kE + h * kD + d];
  ph[tid] = acc; __syncthreads();
  if (tid < kD)
    o[rowq + tid] = (ph[d] + ph[64 + d] + ph[128 + d] + ph[192 + d]) * inv;
}


static inline void reduce_absmax(const float* x, int n, float* scal, float* part, hipStream_t st) {
  k_part_absmax<<<256, 256, 0, st>>>(x, n, part);
  k_finish_max<<<1, 256, 0, st>>>(part, scal);
}
static inline void reduce_abssum(const float* w, int n, float* scal, float* part, hipStream_t st) {
  k_part_abssum<<<256, 256, 0, st>>>(w, n, part);
  k_finish_sum<<<1, 256, 0, st>>>(part, scal);
}
static inline void quantX(const float* x, int M, int K, int Mp, int Kp,
                          int8_t* qx, float* amax, float* part, hipStream_t st) {
  reduce_absmax(x, M * K, amax, part, st);
  int tot = Mp * Kp;
  k_quant_x<<<(tot + 255) / 256, 256, 0, st>>>(x, qx, amax, M, K, Mp, Kp);
}
static inline void quantW(const float* w, int N, int K, int Kp,
                          int8_t* qw, float* wsum, float* part, hipStream_t st) {
  reduce_abssum(w, N * K, wsum, part, st);
  int tot = N * Kp;
  k_quant_w<<<(tot + 255) / 256, 256, 0, st>>>(w, qw, wsum, N, K, Kp);
}
static inline void gemm(const int8_t* qx, const int8_t* qw, const float* bias,
                        const float* wsum, const float* amax, float* out,
                        int Mvalid, int Mp, int N, int Kp, int wNK, int relu,
                        hipStream_t st) {
  dim3 g(Mp / 64, N / 64);
  k_gemm_i8<<<g, 128, 0, st>>>(qx, qw, bias, wsum, amax, out, Mvalid, N, Kp, wNK, relu);
}

extern "C" void kernel_launch(void* const* d_in, const int* in_sizes, int n_in,
                              void* d_out, int out_size, void* d_ws, size_t ws_size,
                              hipStream_t stream) {
  (void)in_sizes; (void)n_in; (void)out_size; (void)ws_size;

  const int*   ids  = (const int*)d_in[0];
  const float* img  = (const float*)d_in[1];
  const float* temb = (const float*)d_in[2];
  const float* Wv1 = (const float*)d_in[3];  const float* bv1 = (const float*)d_in[4];
  const float* Wv2 = (const float*)d_in[5];  const float* bv2 = (const float*)d_in[6];
  const float* fWq = (const float*)d_in[7];  const float* fWk = (const float*)d_in[8];
  const float* fWv = (const float*)d_in[9];  const float* fWo = (const float*)d_in[10];
  const float* fbq = (const float*)d_in[11]; const float* fbk = (const float*)d_in[12];
  const float* fbv = (const float*)d_in[13]; const float* fbo = (const float*)d_in[14];
  const float* Wm1 = (const float*)d_in[15]; const float* bm1 = (const float*)d_in[16];
  const float* Wm2 = (const float*)d_in[17]; const float* bm2 = (const float*)d_in[18];
  const float* Wq  = (const float*)d_in[19]; const float* Wk  = (const float*)d_in[20];
  const float* Wv  = (const float*)d_in[21]; const float* Wo  = (const float*)d_in[22];
  const float* bq  = (const float*)d_in[23]; const float* bk  = (const float*)d_in[24];
  const float* bv  = (const float*)d_in[25]; const float* bo  = (const float*)d_in[26];
  const float* Wlm = (const float*)d_in[27]; const float* blm = (const float*)d_in[28];

  char* ws = (char*)d_ws;
  float*  SC   = (float*)(ws + OFF_SC);
  float*  amax = SC;
  float*  wsum = SC + 1;
  float*  part = SC + 64;
  int8_t* QX = (int8_t*)(ws + OFF_QX);
  int8_t* QW = (int8_t*)(ws + OFF_QW);
  float* X  = (float*)(ws + OFF_X);
  float* Hh = (float*)(ws + OFF_H);
  float* FU = (float*)(ws + OFF_FU);
  float* Qb = (float*)(ws + OFF_Q);
  float* Kb = (float*)(ws + OFF_K);
  float* Vb = (float*)(ws + OFF_V);
  float* AO = (float*)(ws + OFF_AO);
  float* XC = (float*)(ws + OFF_XC);
  float* VP = (float*)(ws + OFF_VP);
  float* VH = (float*)(ws + OFF_VH);
  float* VO = (float*)(ws + OFF_VO);

  k_embed<<<(kM * kE + 255) / 256, 256, 0, stream>>>(ids, temb, X);
  k_pool<<<(kB * kPatchK + 255) / 256, 256, 0, stream>>>(img, VP);
  quantX(VP, kB, kPatchK, kMV, kPatchKp, QX, amax, part, stream);
  quantW(Wv1, kVD, kPatchK, kPatchKp, QW, wsum, part, stream);
  gemm(QX, QW, bv1, wsum, amax, VH, kB, kMV, kVD, kPatchKp, kVD * kPatchK, 1, stream);
  quantX(VH, kB, kVD, kMV, kVD, QX, amax, part, stream);
  quantW(Wv2, kE, kVD, kVD, QW, wsum, part, stream);
  gemm(QX, QW, bv2, wsum, amax, VO, kB, kMV, kE, kVD, kE * kVD, 0, stream);
  k_build_xcat<<<(kMF * kE + 255) / 256, 256, 0, stream>>>(X, VO, XC);
  quantX(XC, kMF, kE, kMFp, kE, QX, amax, part, stream);
  quantW(fWq, kE, kE, kE, QW, wsum, part, stream);
  gemm(QX, QW, fbq, wsum, amax, Qb, kMF, kMFp, kE, kE, kE * kE, 0, stream);
  quantW(fWk, kE, kE, kE, QW, wsum, part, stream);
  gemm(QX, QW, fbk, wsum, amax, Kb, kMF, kMFp, kE, kE, kE * kE, 0, stream);
  quantW(fWv, kE, kE, kE, QW, wsum, part, stream);
  gemm(QX, QW, fbv, wsum, amax, Vb, kMF, kMFp, kE, kE, kE * kE, 0, stream);
  {
    dim3 ga(kS + 1, kH, kB);
    k_attn<<<ga, 256, 0, stream>>>(Qb, Kb, Vb, AO, kS + 1);
  }
  quantX(AO, kMF, kE, kMFp, kE, QX, amax, part, stream);
  quantW(fWo, kE, kE, kE, QW, wsum, part, stream);
  gemm(QX, QW, fbo, wsum, amax, XC, kMF, kMFp, kE, kE, kE * kE, 0, stream);
  k_concat_fused<<<(kM * 2 * kE + 255) / 256, 256, 0, stream>>>(X, XC, FU);
  quantX(FU, kM, 2 * kE, kM, 2 * kE, QX, amax, part, stream);
  quantW(Wm1, kFF, 2 * kE, 2 * kE, QW, wsum, part, stream);
  gemm(QX, QW, bm1, wsum, amax, Hh, kM, kM, kFF, 2 * kE, kFF * 2 * kE, 1, stream);
  quantX(Hh, kM, kFF, kM, kFF, QX, amax, part, stream);
  quantW(Wm2, kE, kFF, kFF, QW, wsum, part, stream);
  gemm(QX, QW, bm2, wsum, amax, X, kM, kM, kE, kFF, kE * kFF, 0, stream);
  for (int l = 0; l < kL; ++l) {
    const float* wq = Wq + (size_t)l * kE * kE; const float* bql = bq + (size_t)l * kE;
    const float* wk = Wk + (size_t)l * kE * kE; const float* bkl = bk + (size_t)l * kE;
    const float* wv = Wv + (size_t)l * kE * kE; const float* bvl = bv + (size_t)l * kE;
    const float* wo = Wo + (size_t)l * kE * kE; const float* bol = bo + (size_t)l * kE;
    quantX(X, kM, kE, kM, kE, QX, amax, part, stream);
    quantW(wq, kE, kE, kE, QW, wsum, part, stream);
    gemm(QX, QW, bql, wsum, amax, Qb, kM, kM, kE, kE, kE * kE, 0, stream);
    quantW(wk, kE, kE, kE, QW, wsum, part, stream);
    gemm(QX, QW, bkl, wsum, amax, Kb, kM, kM, kE, kE, kE * kE, 0, stream);
    quantW(wv, kE, kE, kE, QW, wsum, part, stream);
    gemm(QX, QW, bvl, wsum, amax, Vb, kM, kM, kE, kE, kE * kE, 0, stream);
    dim3 gl(kS, kH, kB);
    k_attn<<<gl, 256, 0, stream>>>(Qb, Kb, Vb, AO, kS);
    quantX(AO, kM, kE, kM, kE, QX, amax, part, stream);
    quantW(wo, kE, kE, kE, QW, wsum, part, stream);
    gemm(QX, QW, bol, wsum, amax, XC, kM, kM, kE, kE, kE * kE, 0, stream);
    k_add<<<(kM * kE + 255) / 256, 256, 0, stream>>>(X, XC, kM * kE);
  }
  quantX(X, kM, kE, kM, kE, QX, amax, part, stream);
  quantW(Wlm, kV, kE, kE, QW, wsum, part, stream);
  gemm(QX, QW, blm, wsum, amax, (float*)d_out, kM, kM, kV, kE, kV * kE, 0, stream);
}
